// TK_v2_1503238554335
// MI455X (gfx1250) — matrix operand order repaired and checked by reading, not compiled, not run on hardware
//
#include <hip/hip_runtime.h>
#include <hip/hip_bf16.h>

// ---------------------------------------------------------------------------
// TK-v2 forward for MI455X (gfx1250, wave32).
// All dense GEMMs via v_wmma_f32_16x16x32_f16 with fully unconditional,
// 16B-vectorized operand loads (padded strides + pre-transposed f16 weights).
// ---------------------------------------------------------------------------

typedef __attribute__((ext_vector_type(16))) _Float16 v16h;
typedef __attribute__((ext_vector_type(8)))  _Float16 h8;
typedef __attribute__((ext_vector_type(8)))  float    v8f;

#define E_DIM  300
#define E_PAD  320
#define FF_DIM 100
#define FF_PAD 128
#define CB_DIM 96
#define AT_DIM 32
#define NHEAD  8
#define B_SZ   256
#define TQ     30
#define TD     200
#define NKER   11
#define DN_NP  208          // 13 n-tiles for the 30x200 cosine matrix

__constant__ float c_mu[NKER]     = {1.0f,0.9f,0.7f,0.5f,0.3f,0.1f,-0.1f,-0.3f,-0.5f,-0.7f,-0.9f};
__constant__ float c_inv2s2[NKER] = {500000.0f,50.f,50.f,50.f,50.f,50.f,50.f,50.f,50.f,50.f,50.f};
__constant__ int   c_wsz[7] = {20,30,50,80,100,120,150};
__constant__ int   c_mw[7]  = {10,7,4,3,2,2,2};
__constant__ int   c_off[8] = {0,10,17,21,24,26,28,30};

static inline int ceil_div(int a, int b) { return (a + b - 1) / b; }

// ---------------------------------------------------------------------------
// Weight prep: W[K,N] f32 row-major -> Wt[Np,Kp] f16 (transposed, zero padded)
// ---------------------------------------------------------------------------
__global__ void prep_weight_kernel(const float* __restrict__ W, _Float16* __restrict__ Wt,
                                   int K, int N, int Kp, int Np)
{
  int idx = blockIdx.x * blockDim.x + threadIdx.x;
  if (idx >= Np * Kp) return;
  int n = idx / Kp, k = idx % Kp;
  float v = (k < K && n < N) ? W[(size_t)k * N + n] : 0.f;
  Wt[idx] = (_Float16)v;
}

// ---------------------------------------------------------------------------
// C[M,Np] = act(A[M,K] @ Wt^T + bias)  ; A f32 (row stride K, K%32==0),
// Wt f16 [Np,K] (n-major, k-contiguous), C f32 (row stride ldc=Np).
// Pad columns (n>=N) are stored as 0 so C can feed the next GEMM's padded K.
// 4 waves / block, one 16x16 tile per wave.  All loads unconditional.
// Fragment layouts per CDNA5 ISA 7.12.2 (f16 16x16x32):
//   A: lane L -> m = L&15, k = k0 + 8*(L>>4) + (j<8 ? j : j+8)
//   B: lane L -> n = L&15, k = k0 + 8*(L>>4) + (j<8 ? j : j+8)
//   C: lane L -> n = L&15, vgpr r -> m = 8*(L>>4) + r
// ---------------------------------------------------------------------------
__global__ void __launch_bounds__(128)
gemm_bias_wmma(const float* __restrict__ A, const _Float16* __restrict__ Wt,
               const float* __restrict__ bias, float* __restrict__ C,
               int N, int Np, int K, int ldc, int relu)
{
  const int lane = threadIdx.x;
  const int n0 = (blockIdx.x * 4 + threadIdx.y) * 16;
  if (n0 >= Np) return;                        // wave-uniform: EXEC stays full
  const int m0 = blockIdx.y * 16;
  const int lo = lane & 15;
  const int hi = lane >> 4;
  const float*    arow = A  + (size_t)(m0 + lo) * K;
  const _Float16* brow = Wt + (size_t)(n0 + lo) * K;
  v8f acc = {0.f,0.f,0.f,0.f,0.f,0.f,0.f,0.f};
  for (int k0 = 0; k0 < K; k0 += 32) {
    const int ka = k0 + (hi << 3);
    const int kb = k0 + (hi << 3);
    float4 a0 = *(const float4*)(arow + ka);
    float4 a1 = *(const float4*)(arow + ka + 4);
    float4 a2 = *(const float4*)(arow + ka + 16);
    float4 a3 = *(const float4*)(arow + ka + 20);
    h8 b0 = *(const h8*)(brow + kb);
    h8 b1 = *(const h8*)(brow + kb + 16);
    v16h a, b;
    a[0]=(_Float16)a0.x;  a[1]=(_Float16)a0.y;  a[2]=(_Float16)a0.z;  a[3]=(_Float16)a0.w;
    a[4]=(_Float16)a1.x;  a[5]=(_Float16)a1.y;  a[6]=(_Float16)a1.z;  a[7]=(_Float16)a1.w;
    a[8]=(_Float16)a2.x;  a[9]=(_Float16)a2.y;  a[10]=(_Float16)a2.z; a[11]=(_Float16)a2.w;
    a[12]=(_Float16)a3.x; a[13]=(_Float16)a3.y; a[14]=(_Float16)a3.z; a[15]=(_Float16)a3.w;
#pragma unroll
    for (int j = 0; j < 8; ++j) { b[j] = b0[j]; b[8 + j] = b1[j]; }
    acc = __builtin_amdgcn_wmma_f32_16x16x32_f16(false, a, false, b, (short)0, acc, false, false);
  }
  const int bn = n0 + lo;
  const int mb = m0 + (hi << 3);
  const float bv = bias[bn < N ? bn : (N - 1)];
  const bool live = (bn < N);
#pragma unroll
  for (int r = 0; r < 8; ++r) {
    float v = live ? (acc[r] + bv) : 0.f;
    if (relu) v = fmaxf(v, 0.f);
    C[(size_t)(mb + r) * ldc + bn] = v;
  }
}

// ---------------------------------------------------------------------------
// cos[b,q,d] = qn[b,q,:] . dn[b,d,:]  (f16 rows, stride E_PAD, zero padded).
// Both operands k-contiguous -> pure 16B loads.  Over-read rows/cols land in
// adjacent ws regions and their outputs are never stored.
// ---------------------------------------------------------------------------
__global__ void __launch_bounds__(128)
cosine_wmma(const _Float16* __restrict__ QN, const _Float16* __restrict__ DN,
            float* __restrict__ COS)
{
  const int lane = threadIdx.x;
  const int n0 = (blockIdx.x * 4 + threadIdx.y) * 16;
  if (n0 >= DN_NP) return;
  const int m0 = blockIdx.y * 16;
  const int b  = blockIdx.z;
  const int lo = lane & 15;
  const int hi = lane >> 4;
  const _Float16* q = QN + ((size_t)b * TQ + (m0 + lo)) * E_PAD;
  const _Float16* d = DN + ((size_t)b * TD + (n0 + lo)) * E_PAD;
  v8f acc = {0.f,0.f,0.f,0.f,0.f,0.f,0.f,0.f};
  for (int k0 = 0; k0 < E_PAD; k0 += 32) {
    const int ka = k0 + (hi << 3);
    const int kb = k0 + (hi << 3);
    h8 qa0 = *(const h8*)(q + ka);
    h8 qa1 = *(const h8*)(q + ka + 16);
    h8 db0 = *(const h8*)(d + kb);
    h8 db1 = *(const h8*)(d + kb + 16);
    v16h a, bf;
#pragma unroll
    for (int j = 0; j < 8; ++j) {
      a[j] = qa0[j]; a[8 + j] = qa1[j];
      bf[j] = db0[j]; bf[8 + j] = db1[j];
    }
    acc = __builtin_amdgcn_wmma_f32_16x16x32_f16(false, a, false, bf, (short)0, acc, false, false);
  }
  const int bn = n0 + lo;
  const int mb = m0 + (hi << 3);
  if (bn < TD) {
#pragma unroll
    for (int r = 0; r < 8; ++r) {
      int m = mb + r;
      if (m < TQ) COS[((size_t)b * TQ + m) * TD + bn] = acc[r];
    }
  }
}

// ---------------------------------------------------------------------------
// x[b,t,e] = emb*mask + sinusoid(t,e), stride E_PAD, pad columns = 0.
// ---------------------------------------------------------------------------
__global__ void pos_add_kernel(const float* __restrict__ emb, const float* __restrict__ mask,
                               float* __restrict__ out, int T, int total)
{
  int idx = blockIdx.x * blockDim.x + threadIdx.x;
  if (idx >= total) return;
  int e   = idx % E_PAD;
  int row = idx / E_PAD;
  int t   = row % T;
  float v = 0.f;
  if (e < E_DIM) {
    const float cexp = 0.06181436365f;        // ln(10000)/149
    float sv;
    if (e < 150) sv = sinf((float)t * expf(-(float)e * cexp));
    else         sv = cosf((float)t * expf(-(float)(e - 150) * cexp));
    v = emb[(size_t)row * E_DIM + e] * mask[row] + sv;
  }
  out[idx] = v;
}

// ---------------------------------------------------------------------------
// Out = LN(X + R) * g + b over 300 dims; rows have stride E_PAD (pads
// untouched and stay zero).  One wave32 per row.
// ---------------------------------------------------------------------------
__global__ void resid_ln_kernel(const float* __restrict__ X, const float* __restrict__ R,
                                const float* __restrict__ g, const float* __restrict__ bb,
                                float* __restrict__ Out, int rows)
{
  int gw   = (blockIdx.x * blockDim.x + threadIdx.x) >> 5;
  int lane = threadIdx.x & 31;
  if (gw >= rows) return;
  const float* x = X + (size_t)gw * E_PAD;
  const float* r = R + (size_t)gw * E_PAD;
  float s = 0.f;
  for (int e = lane; e < E_DIM; e += 32) s += x[e] + r[e];
#pragma unroll
  for (int o = 16; o > 0; o >>= 1) s += __shfl_xor(s, o, 32);
  float mean = s * (1.f / 300.f);
  float v = 0.f;
  for (int e = lane; e < E_DIM; e += 32) { float d = x[e] + r[e] - mean; v += d * d; }
#pragma unroll
  for (int o = 16; o > 0; o >>= 1) v += __shfl_xor(v, o, 32);
  float inv = 1.f / (sqrtf(v * (1.f / 300.f)) + 1e-6f);
  float* o = Out + (size_t)gw * E_PAD;
  for (int e = lane; e < E_DIM; e += 32)
    o[e] = g[e] * ((x[e] + r[e]) - mean) * inv + bb[e];
}

// ---------------------------------------------------------------------------
// Masked MHSA core, one block per (batch, head), dh = 4.
// out_d = sum_j e_j*m_j*v_jd / (sum_j e_j*m_j + 1e-13*sum_j e_j)
// ---------------------------------------------------------------------------
__global__ void attention_kernel(const float* __restrict__ comb, const float* __restrict__ mask,
                                 float* __restrict__ att, int T)
{
  const int b = blockIdx.x;
  const int h = blockIdx.y;
  __shared__ float kv[TD * 8];
  const float* cb = comb + (size_t)b * T * CB_DIM;
  for (int idx = threadIdx.x; idx < T * 4; idx += blockDim.x) {
    int j = idx >> 2, d = idx & 3;
    kv[j * 8 + d]     = cb[(size_t)j * CB_DIM + 32 + h * 4 + d];
    kv[j * 8 + 4 + d] = cb[(size_t)j * CB_DIM + 64 + h * 4 + d];
  }
  __syncthreads();
  const float invs = 0.1643989873f;            // 1/sqrt(300/8)
  for (int i = threadIdx.x; i < T; i += blockDim.x) {
    float q0 = cb[(size_t)i * CB_DIM + h * 4 + 0];
    float q1 = cb[(size_t)i * CB_DIM + h * 4 + 1];
    float q2 = cb[(size_t)i * CB_DIM + h * 4 + 2];
    float q3 = cb[(size_t)i * CB_DIM + h * 4 + 3];
    float mmax = -3.4e38f;
    for (int j = 0; j < T; ++j) {
      float sdot = (q0 * kv[j*8] + q1 * kv[j*8+1] + q2 * kv[j*8+2] + q3 * kv[j*8+3]) * invs;
      mmax = fmaxf(mmax, sdot * mask[(size_t)b * T + j]);
    }
    float denomE = 0.f, S = 0.f, a0 = 0.f, a1 = 0.f, a2 = 0.f, a3 = 0.f;
    for (int j = 0; j < T; ++j) {
      float mj = mask[(size_t)b * T + j];
      float sdot = (q0 * kv[j*8] + q1 * kv[j*8+1] + q2 * kv[j*8+2] + q3 * kv[j*8+3]) * invs;
      float e = expf(sdot * mj - mmax);
      denomE += e;
      float em = e * mj;
      S += em;
      a0 += em * kv[j*8+4]; a1 += em * kv[j*8+5]; a2 += em * kv[j*8+6]; a3 += em * kv[j*8+7];
    }
    float inv = 1.f / (S + 1e-13f * denomE);
    float* o = att + ((size_t)b * T + i) * AT_DIM + h * 4;
    o[0] = a0 * inv; o[1] = a1 * inv; o[2] = a2 * inv; o[3] = a3 * inv;
  }
}

// ---------------------------------------------------------------------------
// q = (mix*(emb*mask) + (1-mix)*enc)*mask ; out = f16(q/(||q||+1e-13)),
// row stride E_PAD on enc/out (pad halves written as 0).  emb stride 300.
// ---------------------------------------------------------------------------
__global__ void mixnorm_kernel(const float* __restrict__ emb, const float* __restrict__ enc,
                               const float* __restrict__ mask, const float* __restrict__ mixer,
                               _Float16* __restrict__ outh, int rows)
{
  int gw   = (blockIdx.x * blockDim.x + threadIdx.x) >> 5;
  int lane = threadIdx.x & 31;
  if (gw >= rows) return;
  float mix = mixer[0];
  float mk  = mask[gw];
  const float* a = emb + (size_t)gw * E_DIM;
  const float* c = enc + (size_t)gw * E_PAD;
  float ss = 0.f;
  for (int e = lane; e < E_DIM; e += 32) {
    float v = (mix * a[e] * mk + (1.f - mix) * c[e]) * mk;
    ss += v * v;
  }
#pragma unroll
  for (int o = 16; o > 0; o >>= 1) ss += __shfl_xor(ss, o, 32);
  float inv = 1.f / (sqrtf(ss) + 1e-13f);
  for (int e = lane; e < E_PAD; e += 32) {
    float v = (e < E_DIM) ? (mix * a[e] * mk + (1.f - mix) * c[e]) * mk * inv : 0.f;
    outh[(size_t)gw * E_PAD + e] = (_Float16)v;
  }
}

// ---------------------------------------------------------------------------
// Fused kernel-pooling + window scoring + merge.  One block per batch.
// ---------------------------------------------------------------------------
struct ScoreParams {
  const float* kw[7];
  const float* scaler[7];
  const float* wsc[7];
  const float* merger;
};

__global__ void score_kernel(const float* __restrict__ COS, const float* __restrict__ qmask,
                             const float* __restrict__ dmask, ScoreParams sp,
                             float* __restrict__ out)
{
  const int b = blockIdx.x;
  __shared__ float perk[30][NKER];
  __shared__ float cfg_score[7];
  for (int i = threadIdx.x; i < 30 * NKER; i += blockDim.x) (&perk[0][0])[i] = 0.f;
  __syncthreads();

  for (int t = threadIdx.x; t < TQ * 30; t += blockDim.x) {
    int q    = t / 30;
    int slot = t % 30;
    int cfg  = 0;
    while (slot >= c_off[cfg + 1]) cfg++;
    int j  = slot - c_off[cfg];
    int w  = c_wsz[cfg];
    int d0 = j * w;
    int d1 = d0 + w; if (d1 > TD) d1 = TD;
    float pk[NKER];
#pragma unroll
    for (int k = 0; k < NKER; ++k) pk[k] = 0.f;
    if (d0 < TD) {
      float qm = qmask[(size_t)b * TQ + q];
      const float* crow = COS + ((size_t)b * TQ + q) * TD;
      for (int d = d0; d < d1; ++d) {
        float m = qm * dmask[(size_t)b * TD + d];
        float c = tanhf(crow[d] * m);
#pragma unroll
        for (int k = 0; k < NKER; ++k) {
          float df = c - c_mu[k];
          pk[k] += expf(-df * df * c_inv2s2[k]) * m;
        }
      }
    }
    float s = 0.f;
#pragma unroll
    for (int k = 0; k < NKER; ++k) s += pk[k];
    if (s != 0.f) {
#pragma unroll
      for (int k = 0; k < NKER; ++k)
        atomicAdd(&perk[slot][k], logf(fmaxf(pk[k], 1e-10f)));
    }
  }
  __syncthreads();

  if (threadIdx.x < 7) {
    int cfg = threadIdx.x;
    int mw  = c_mw[cfg];
    float wse[10];
    for (int j = 0; j < mw; ++j) {
      float ws = 0.f;
      for (int k = 0; k < NKER; ++k) ws += perk[c_off[cfg] + j][k] * sp.kw[cfg][k];
      wse[j] = (ws != 0.f) ? expf(ws * sp.scaler[cfg][0]) : 0.f;
    }
    for (int a = 1; a < mw; ++a) {
      float v = wse[a]; int p = a - 1;
      while (p >= 0 && wse[p] < v) { wse[p + 1] = wse[p]; --p; }
      wse[p + 1] = v;
    }
    float sc = 0.f;
    for (int j = 0; j < mw; ++j) sc += wse[j] * sp.wsc[cfg][j];
    cfg_score[cfg] = sc;
  }
  __syncthreads();
  if (threadIdx.x == 0) {
    float f = 0.f;
    for (int i = 0; i < 7; ++i) f += cfg_score[i] * sp.merger[i];
    out[b] = f;
  }
}

// ---------------------------------------------------------------------------
// Host-side pipeline
// ---------------------------------------------------------------------------
static void run_encoder(const float* emb, const float* mask, int T,
                        const float* const* P, const _Float16* wbuf,
                        float* bx, float* bff, float* btmp, float* batt,
                        hipStream_t s)
{
  const int rows  = B_SZ * T;
  const int total = rows * E_PAD;
  pos_add_kernel<<<ceil_div(total, 256), 256, 0, s>>>(emb, mask, bx, T, total);
  for (int l = 0; l < 2; ++l) {
    const float* b1   = P[4 + l * 12 + 1];
    const float* b2   = P[4 + l * 12 + 3];
    const float* ffg  = P[4 + l * 12 + 4];
    const float* ffb  = P[4 + l * 12 + 5];
    const float* qkvb = P[4 + l * 12 + 7];
    const float* outb = P[4 + l * 12 + 9];
    const float* lng  = P[4 + l * 12 + 10];
    const float* lnb  = P[4 + l * 12 + 11];
    const _Float16* w1t   = wbuf + (size_t)l * 122880;
    const _Float16* w2t   = w1t + 40960;    // FF_PAD x E_PAD
    const _Float16* qkvt  = w2t + 40960;    // E_PAD  x FF_PAD
    const _Float16* outt  = qkvt + 30720;   // CB_DIM x E_PAD

    gemm_bias_wmma<<<dim3(FF_PAD / 64, rows / 16), dim3(32, 4), 0, s>>>(
        bx, w1t, b1, btmp, FF_DIM, FF_PAD, E_PAD, FF_PAD, 1);
    gemm_bias_wmma<<<dim3(E_PAD / 64, rows / 16), dim3(32, 4), 0, s>>>(
        btmp, w2t, b2, bff, E_DIM, E_PAD, FF_PAD, E_PAD, 0);
    resid_ln_kernel<<<ceil_div(rows * 32, 256), 256, 0, s>>>(bff, bx, ffg, ffb, bff, rows);
    gemm_bias_wmma<<<dim3(ceil_div(CB_DIM, 64), rows / 16), dim3(32, 4), 0, s>>>(
        bff, qkvt, qkvb, btmp, CB_DIM, CB_DIM, E_PAD, CB_DIM, 0);
    attention_kernel<<<dim3(B_SZ, NHEAD), 128, 0, s>>>(btmp, mask, batt, T);
    gemm_bias_wmma<<<dim3(E_PAD / 64, rows / 16), dim3(32, 4), 0, s>>>(
        batt, outt, outb, bx, E_DIM, E_PAD, AT_DIM, E_PAD, 0);
    resid_ln_kernel<<<ceil_div(rows * 32, 256), 256, 0, s>>>(bx, bff, lng, lnb, bx, rows);
  }
}

extern "C" void kernel_launch(void* const* d_in, const int* in_sizes, int n_in,
                              void* d_out, int out_size, void* d_ws, size_t ws_size,
                              hipStream_t stream)
{
  (void)in_sizes; (void)out_size; (void)ws_size;
  const float* P[64];
  for (int i = 0; i < n_in && i < 64; ++i) P[i] = (const float*)d_in[i];
  const float* qe = P[0];
  const float* de = P[1];
  const float* qm = P[2];
  const float* dm = P[3];

  // workspace layout (16B-aligned element offsets)
  float*     bx   = (float*)d_ws;                                    // rows*E_PAD
  float*     bff  = bx   + (size_t)B_SZ * TD * E_PAD;
  float*     btmp = bff  + (size_t)B_SZ * TD * E_PAD;                // rows*FF_PAD
  float*     batt = btmp + (size_t)B_SZ * TD * FF_PAD;               // rows*AT_DIM
  _Float16*  qn   = (_Float16*)(batt + (size_t)B_SZ * TD * AT_DIM);  // B*TQ*E_PAD
  _Float16*  dn   = qn   + (size_t)B_SZ * TQ * E_PAD;                // B*TD*E_PAD
  float*     cosb = (float*)(dn + (size_t)B_SZ * TD * E_PAD);        // B*TQ*TD
  _Float16*  wbuf = (_Float16*)(cosb + (size_t)B_SZ * TQ * TD);      // 2*122880

  // Pre-transpose + pad + f16-convert all GEMM weights.
  for (int l = 0; l < 2; ++l) {
    _Float16* w1t  = wbuf + (size_t)l * 122880;
    _Float16* w2t  = w1t + 40960;
    _Float16* qkvt = w2t + 40960;
    _Float16* outt = qkvt + 30720;
    prep_weight_kernel<<<ceil_div(FF_PAD * E_PAD, 256), 256, 0, stream>>>(
        P[4 + l * 12 + 0], w1t, E_DIM, FF_DIM, E_PAD, FF_PAD);
    prep_weight_kernel<<<ceil_div(E_PAD * FF_PAD, 256), 256, 0, stream>>>(
        P[4 + l * 12 + 2], w2t, FF_DIM, E_DIM, FF_PAD, E_PAD);
    prep_weight_kernel<<<ceil_div(CB_DIM * E_PAD, 256), 256, 0, stream>>>(
        P[4 + l * 12 + 6], qkvt, E_DIM, CB_DIM, E_PAD, CB_DIM);
    prep_weight_kernel<<<ceil_div(E_PAD * AT_DIM, 256), 256, 0, stream>>>(
        P[4 + l * 12 + 8], outt, AT_DIM, E_DIM, AT_DIM, E_PAD);
  }

  // query encoder -> mix/normalize -> f16
  run_encoder(qe, qm, TQ, P, wbuf, bx, bff, btmp, batt, stream);
  mixnorm_kernel<<<ceil_div(B_SZ * TQ * 32, 256), 256, 0, stream>>>(
      qe, bx, qm, P[28], qn, B_SZ * TQ);

  // document encoder -> mix/normalize -> f16
  run_encoder(de, dm, TD, P, wbuf, bx, bff, btmp, batt, stream);
  mixnorm_kernel<<<ceil_div(B_SZ * TD * 32, 256), 256, 0, stream>>>(
      de, bx, dm, P[28], dn, B_SZ * TD);

  // per-batch 30x200 cosine matrix via WMMA
  cosine_wmma<<<dim3(DN_NP / 64 + 1, 2, B_SZ), dim3(32, 4), 0, stream>>>(qn, dn, cosb);

  // fused kernel pooling + window scoring + merge
  ScoreParams sp;
  for (int i = 0; i < 7; ++i) {
    sp.kw[i]     = P[29 + 3 * i];
    sp.scaler[i] = P[30 + 3 * i];
    sp.wsc[i]    = P[31 + 3 * i];
  }
  sp.merger = P[50];
  score_kernel<<<B_SZ, 256, 0, stream>>>(cosb, qm, dm, sp, (float*)d_out);
}
